// DynamicProcessSolver_10033043603706
// MI455X (gfx1250) — matrix operand order repaired and checked by reading, not compiled, not run on hardware
//
#include <hip/hip_runtime.h>
#include <hip/hip_bf16.h>
#include <math.h>

typedef __attribute__((ext_vector_type(16))) _Float16 v16h;
typedef __attribute__((ext_vector_type(8)))  _Float16 v8h;
typedef __attribute__((ext_vector_type(8)))  float    v8f;

#define B_TOT 1024
#define S_TOT 2048
#define F_IN  9
#define NO    8
#define LB    3
#define T_OUT (S_TOT - 1 - LB)   // 2044 output timesteps
#define MT    16                 // batch rows per workgroup (one WMMA M tile)
#define NWG   (B_TOT / MT)       // 64 independent workgroups (recurrence is per-row)
#define NTHREADS 256             // 8 wave32s
#define LN_EPS 0.001f

// LDS K-strides padded by +8 halves => row stride = 16B (mod 32 banks):
// the 16 A-fragment rows land on distinct bank quads (conflict-free ds_load_b128).
#define LDK_XA 72    // 64 + 8
#define LDK_H1 520   // 512 + 8
#define LDK_HH 264   // 256 + 8
#define LDK_CB 296   // 288 + 8
#define LDK_T2 520   // 512 + 8

// ---------------------------------------------------------------------------
// Weight pre-swizzle: fp32 row-major (Ks x Ns) -> f16 B-fragment-major tiles.
// B fragment (16x16x32 f16 WMMA, B is 32x16 KxN):
//   lane n (n<16): column n, halves 0..7 = K 0..7, halves 8..15 = K 16..23 of the tile
//   lane n+16    : column n, halves 0..7 = K 8..15, halves 8..15 = K 24..31 of the tile
// ---------------------------------------------------------------------------
__global__ void swizzle_weight(const float* __restrict__ src, _Float16* __restrict__ dst,
                               int Ks, int Ns, int KT, int NT) {
    size_t idx = (size_t)blockIdx.x * blockDim.x + threadIdx.x;
    size_t total = (size_t)KT * NT * 512;
    if (idx >= total) return;
    int h    = (int)(idx & 15);
    int lane = (int)((idx >> 4) & 31);
    size_t tile = idx >> 9;
    int nt = (int)(tile % NT);
    int kt = (int)(tile / NT);
    int k = kt * 32 + ((lane >> 4) & 1) * 8 + (h & 7) + (h >> 3) * 16;
    int n = nt * 16 + (lane & 15);
    float v = (k < Ks && n < Ns) ? src[(size_t)k * Ns + n] : 0.0f;
    dst[idx] = (_Float16)v;
}

// ---------------------------------------------------------------------------
// Fragment helpers
// ---------------------------------------------------------------------------
// A fragment (16x32 f16) from LDS row-major [16][ldk] halves (ldk % 8 == 0).
// ISA layout: lane m (g = lane>>4): halves 0..7 = K g*8.., halves 8..15 = K 16+g*8..
__device__ inline v16h lds_load_a(const _Float16* base, int ldk, int kt) {
    int lane = threadIdx.x & 31;
    int m = lane & 15, g = (lane >> 4) & 1;
    const _Float16* row = base + m * ldk + kt * 32 + g * 8;
    v8h lo = *(const v8h*)(row);
    v8h hi = *(const v8h*)(row + 16);
    v16h r;
#pragma unroll
    for (int i = 0; i < 8; ++i) { r[i] = lo[i]; r[i + 8] = hi[i]; }
    return r;
}

// B fragment from pre-swizzled global weights: one 32B contiguous load per lane.
__device__ inline v16h glb_load_b(const _Float16* __restrict__ wbase, int NT, int kt, int nt) {
    int lane = threadIdx.x & 31;
    const _Float16* p = wbase + (((size_t)kt * NT + nt) * 32 + lane) * 16;
    return *(const v16h*)p;
}

__device__ inline v8f wmma_step(v16h a, v16h b, v8f c) {
    return __builtin_amdgcn_wmma_f32_16x16x32_f16(false, a, false, b, (short)0, c, false, false);
}

// Branch-free tanh: native v_tanh_f32 if the builtin exists, else exp2+rcp.
__device__ inline float fast_tanh(float x) {
#if defined(__has_builtin)
#if __has_builtin(__builtin_amdgcn_tanhf)
    return __builtin_amdgcn_tanhf(x);
#else
    float cx = fminf(fmaxf(x, -10.0f), 10.0f);
    float e  = __builtin_amdgcn_exp2f(cx * 2.8853900817779268f); // e^{2x}
    return 1.0f - 2.0f * __builtin_amdgcn_rcpf(e + 1.0f);
#endif
#else
    float cx = fminf(fmaxf(x, -10.0f), 10.0f);
    float e  = __builtin_amdgcn_exp2f(cx * 2.8853900817779268f);
    return 1.0f - 2.0f * __builtin_amdgcn_rcpf(e + 1.0f);
#endif
}

// ---------------------------------------------------------------------------
// Persistent per-workgroup scan kernel: 16 batch rows, all 2044 timesteps.
// ---------------------------------------------------------------------------
__global__ __launch_bounds__(NTHREADS) void solver_kernel(
    const float* __restrict__ inputs,
    const float* __restrict__ ln1_g, const float* __restrict__ ln1_b,
    const _Float16* __restrict__ w1s, const float* __restrict__ b1,
    const _Float16* __restrict__ w2s, const float* __restrict__ b2,
    const _Float16* __restrict__ ups, const float* __restrict__ updb,
    const float* __restrict__ ln2_g, const float* __restrict__ ln2_b,
    const float* __restrict__ betaw,
    const _Float16* __restrict__ d1s, const float* __restrict__ db1,
    const _Float16* __restrict__ d2s, const float* __restrict__ db2,
    const int* __restrict__ fixp,
    float* __restrict__ out)
{
    __shared__ alignas(16) _Float16 sXA[16 * LDK_XA];
    __shared__ alignas(16) unsigned char sH1raw[16 * LDK_H1 * 2]; // f16 h1 / fp32 st_pre
    __shared__ alignas(16) _Float16 sHH[16 * LDK_HH];
    __shared__ alignas(16) _Float16 sCB[16 * LDK_CB];
    __shared__ alignas(16) _Float16 sT2[16 * LDK_T2];
    __shared__ float sWin[16 * LB * NO];
    __shared__ float sEct[16];
    __shared__ float sY[16 * NO];

    _Float16* sH1  = (_Float16*)sH1raw;
    float*    sSTF = (float*)sH1raw;     // 16*256 fp32 fits in 16*LDK_H1*2 bytes

    const int tid  = threadIdx.x;
    const int lane = tid & 31;
    const int wv   = tid >> 5;            // wave 0..7
    const int col  = lane & 15;
    const int gg   = (lane >> 4) & 1;
    const int b0   = blockIdx.x * MT;
    const int fix0 = fixp[0], fix1 = fixp[1];

    // One-time: zero K-pad regions, load initial window.
    for (int idx = tid; idx < 16 * LDK_XA; idx += NTHREADS) sXA[idx] = (_Float16)0.f;
    for (int idx = tid; idx < 16 * LDK_CB; idx += NTHREADS) sCB[idx] = (_Float16)0.f;
    for (int idx = tid; idx < 16 * LB * NO; idx += NTHREADS) {
        int b = idx / (LB * NO); int rem = idx % (LB * NO);
        int r = rem / NO; int c = rem % NO;
        sWin[idx] = inputs[(((size_t)(b0 + b)) * S_TOT + r) * F_IN + 1 + c];
    }
    __syncthreads();

    for (int i = LB; i <= S_TOT - 2; ++i) {
        // ---- Phase A: x = [lc | win], LayerNorm over 11 -> sXA f16 ----
        if (tid < 48) {
            int b = tid / 3, r = tid % 3;
            int s = i - LB + r;
            size_t base = (((size_t)(b0 + b)) * S_TOT + s) * F_IN;
            float l0 = inputs[base];
            float lm = (s > 0) ? inputs[base - F_IN] : 0.f;
            float v[11];
            v[0] = l0;                        // loading
            v[1] = l0;                        // cumsum over singleton axis == loading
            v[2] = (s > 0) ? (l0 - lm) : 0.f; // padded diff
#pragma unroll
            for (int c = 0; c < NO; ++c) v[3 + c] = sWin[(b * LB + r) * NO + c];
            float mean = 0.f;
#pragma unroll
            for (int d = 0; d < 11; ++d) mean += v[d];
            mean *= (1.f / 11.f);
            float var = 0.f;
#pragma unroll
            for (int d = 0; d < 11; ++d) { float dd = v[d] - mean; var += dd * dd; }
            var *= (1.f / 11.f);
            float rs = rsqrtf(var + LN_EPS);
#pragma unroll
            for (int d = 0; d < 11; ++d)
                sXA[b * LDK_XA + r * 11 + d] = (_Float16)((v[d] - mean) * rs * ln1_g[d] + ln1_b[d]);
        }
        __syncthreads();

        // ---- Phase B: h1 = tanh(x @ enc_w1 + b1)   [16x64]@[64x512] ----
#pragma unroll 1
        for (int q = 0; q < 4; ++q) {
            int nt = wv * 4 + q;
            v8f acc = {};
#pragma unroll
            for (int kt = 0; kt < 2; ++kt)
                acc = wmma_step(lds_load_a(sXA, LDK_XA, kt), glb_load_b(w1s, 32, kt, nt), acc);
            float bias = b1[nt * 16 + col];
#pragma unroll
            for (int r = 0; r < 8; ++r)
                sH1[(r + gg * 8) * LDK_H1 + nt * 16 + col] = (_Float16)fast_tanh(acc[r] + bias);
        }
        __syncthreads();

        // ---- Phase C: h = h1 @ enc_w2 + b2   [16x512]@[512x256] ----
#pragma unroll 1
        for (int q = 0; q < 2; ++q) {
            int nt = wv * 2 + q;
            v8f acc = {};
#pragma unroll 2
            for (int kt = 0; kt < 16; ++kt)
                acc = wmma_step(lds_load_a(sH1, LDK_H1, kt), glb_load_b(w2s, 16, kt, nt), acc);
            float bias = b2[nt * 16 + col];
#pragma unroll
            for (int r = 0; r < 8; ++r)
                sHH[(r + gg * 8) * LDK_HH + nt * 16 + col] = (_Float16)(acc[r] + bias);
        }
        __syncthreads();

        // ---- Phase D: st_pre = h @ upd_w + upd_b (fp32, reuses sH1); ect = h @ beta_w ----
#pragma unroll 1
        for (int q = 0; q < 2; ++q) {
            int nt = wv * 2 + q;
            v8f acc = {};
#pragma unroll 2
            for (int kt = 0; kt < 8; ++kt)
                acc = wmma_step(lds_load_a(sHH, LDK_HH, kt), glb_load_b(ups, 16, kt, nt), acc);
            float bias = updb[nt * 16 + col];
#pragma unroll
            for (int r = 0; r < 8; ++r)
                sSTF[(r + gg * 8) * 256 + nt * 16 + col] = acc[r] + bias;
        }
        // ect: wave-parallel dot(h[b,:], beta_w) — wave w owns rows 2w, 2w+1
#pragma unroll
        for (int rr = 0; rr < 2; ++rr) {
            int b = wv * 2 + rr;
            float e = 0.f;
#pragma unroll
            for (int j = 0; j < 8; ++j) {
                int k = lane + j * 32;
                e += (float)sHH[b * LDK_HH + k] * betaw[k];
            }
#pragma unroll
            for (int off = 16; off > 0; off >>= 1) e += __shfl_xor(e, off, 32);
            if (lane == 0) sEct[b] = e;
        }
        __syncthreads();

        // ---- Phase E: LN2 over 256 per row -> comb f16; comb[256] = ect ----
#pragma unroll
        for (int rr = 0; rr < 2; ++rr) {
            int b = wv * 2 + rr;
            float sum = 0.f, sq = 0.f;
#pragma unroll
            for (int j = 0; j < 8; ++j) {
                float vv = sSTF[b * 256 + lane + j * 32];
                sum += vv; sq += vv * vv;
            }
#pragma unroll
            for (int off = 16; off > 0; off >>= 1) {
                sum += __shfl_xor(sum, off, 32);
                sq  += __shfl_xor(sq,  off, 32);
            }
            float mean = sum * (1.f / 256.f);
            float var  = sq * (1.f / 256.f) - mean * mean;
            float rs   = rsqrtf(var + LN_EPS);
#pragma unroll
            for (int j = 0; j < 8; ++j) {
                int k = lane + j * 32;
                float vv = sSTF[b * 256 + k];
                sCB[b * LDK_CB + k] = (_Float16)((vv - mean) * rs * ln2_g[k] + ln2_b[k]);
            }
            if (lane == 0) sCB[b * LDK_CB + 256] = (_Float16)sEct[b];
        }
        __syncthreads();

        // ---- Phase F: t2 = tanh(comb @ dec_w1 + db1)  [16x288]@[288x512] ----
#pragma unroll 1
        for (int q = 0; q < 4; ++q) {
            int nt = wv * 4 + q;
            v8f acc = {};
#pragma unroll 2
            for (int kt = 0; kt < 9; ++kt)
                acc = wmma_step(lds_load_a(sCB, LDK_CB, kt), glb_load_b(d1s, 32, kt, nt), acc);
            float bias = db1[nt * 16 + col];
#pragma unroll
            for (int r = 0; r < 8; ++r)
                sT2[(r + gg * 8) * LDK_T2 + nt * 16 + col] = (_Float16)fast_tanh(acc[r] + bias);
        }
        __syncthreads();

        // ---- Phase G (wave 0): y = t2 @ dec_w2 + db2; fix overrides; emit output ----
        if (wv == 0) {
            v8f acc = {};
#pragma unroll 2
            for (int kt = 0; kt < 16; ++kt)
                acc = wmma_step(lds_load_a(sT2, LDK_T2, kt), glb_load_b(d2s, 1, kt, 0), acc);
            if (col < NO) {
                float bias = db2[col];
#pragma unroll
                for (int r = 0; r < 8; ++r) {
                    int b = r + gg * 8;
                    float val = acc[r] + bias;
                    if (col == fix0)
                        val = inputs[(((size_t)(b0 + b)) * S_TOT + (i + 1)) * F_IN + 1 + fix0];
                    if (col == fix1)
                        val = inputs[(((size_t)(b0 + b)) * S_TOT + (i + 1)) * F_IN + 1 + fix1];
                    sY[b * NO + col] = val;
                    // output layout (NO, B, T)
                    out[((size_t)col * B_TOT + (b0 + b)) * T_OUT + (i - LB)] = val;
                }
            }
        }
        __syncthreads();

        // ---- Phase H: slide window (one thread per (b,c) column: race-free) ----
        if (tid < 16 * NO) {
            int b = tid / NO, c = tid % NO;
            float v1 = sWin[(b * LB + 1) * NO + c];
            float v2 = sWin[(b * LB + 2) * NO + c];
            sWin[(b * LB + 0) * NO + c] = v1;
            sWin[(b * LB + 1) * NO + c] = v2;
            sWin[(b * LB + 2) * NO + c] = sY[b * NO + c];
        }
        __syncthreads();
    }
}

// ---------------------------------------------------------------------------
extern "C" void kernel_launch(void* const* d_in, const int* in_sizes, int n_in,
                              void* d_out, int out_size, void* d_ws, size_t ws_size,
                              hipStream_t stream) {
    (void)in_sizes; (void)n_in; (void)out_size; (void)ws_size;
    const float* inputs = (const float*)d_in[0];
    const float* ln1_g  = (const float*)d_in[1];
    const float* ln1_b  = (const float*)d_in[2];
    const float* enc_w1 = (const float*)d_in[3];
    const float* enc_b1 = (const float*)d_in[4];
    const float* enc_w2 = (const float*)d_in[5];
    const float* enc_b2 = (const float*)d_in[6];
    const float* upd_w  = (const float*)d_in[7];
    const float* upd_b  = (const float*)d_in[8];
    const float* ln2_g  = (const float*)d_in[9];
    const float* ln2_b  = (const float*)d_in[10];
    const float* beta_w = (const float*)d_in[11];
    const float* dec_w1 = (const float*)d_in[12];
    const float* dec_b1 = (const float*)d_in[13];
    const float* dec_w2 = (const float*)d_in[14];
    const float* dec_b2 = (const float*)d_in[15];
    const int*   fix    = (const int*)d_in[16];
    float* out = (float*)d_out;

    _Float16* ws = (_Float16*)d_ws;
    size_t off = 0;
    _Float16* w1s = ws + off; off += (size_t)2  * 32 * 512;  // enc_w1: K 33->64,  N 512
    _Float16* w2s = ws + off; off += (size_t)16 * 16 * 512;  // enc_w2: K 512,     N 256
    _Float16* ups = ws + off; off += (size_t)8  * 16 * 512;  // upd_w : K 256,     N 256
    _Float16* d1s = ws + off; off += (size_t)9  * 32 * 512;  // dec_w1: K 257->288,N 512
    _Float16* d2s = ws + off; off += (size_t)16 * 1  * 512;  // dec_w2: K 512,     N 8->16

    auto launch_sw = [&](const float* src, _Float16* dst, int Ks, int Ns, int KT, int NT) {
        size_t total = (size_t)KT * NT * 512;
        int blocks = (int)((total + 255) / 256);
        swizzle_weight<<<blocks, 256, 0, stream>>>(src, dst, Ks, Ns, KT, NT);
    };
    launch_sw(enc_w1, w1s, 33, 512, 2, 32);
    launch_sw(enc_w2, w2s, 512, 256, 16, 16);
    launch_sw(upd_w,  ups, 256, 256, 8, 16);
    launch_sw(dec_w1, d1s, 257, 512, 9, 32);
    launch_sw(dec_w2, d2s, 512, 8, 16, 1);

    solver_kernel<<<NWG, NTHREADS, 0, stream>>>(
        inputs, ln1_g, ln1_b,
        w1s, enc_b1, w2s, enc_b2, ups, upd_b,
        ln2_g, ln2_b, beta_w,
        d1s, dec_b1, d2s, dec_b2, fix, out);
}
